// SnaXIA_8830452760643
// MI455X (gfx1250) — matrix operand order repaired and checked by reading, not compiled, not run on hardware
//
#include <hip/hip_runtime.h>
#include <hip/hip_bf16.h>

// ---------------------------------------------------------------------------
// Model constants (from reference)
// ---------------------------------------------------------------------------
static constexpr int Bc  = 16;
static constexpr int Sc  = 512;
static constexpr int Hc  = 256;
static constexpr int Vc  = 8000;
static constexpr int NHc = 4;
static constexpr int FFc = 512;
static constexpr int Mc  = 64;
static constexpr int NLc = 2;
static constexpr int HDc = Hc / NHc;          // 64

typedef __attribute__((ext_vector_type(16))) __bf16 v16bf;
typedef __attribute__((ext_vector_type(2)))  __bf16 bf16x2;
typedef __attribute__((ext_vector_type(8)))  float  v8f;

union SnxFrag { v16bf v; unsigned int u[8]; uint4 q[2]; };

// Pack two floats into one dword of 2xbf16 (lo = even k, hi = odd k).
// bf16x2 init + bit_cast folds to a single v_cvt_pk_bf16_f32.
__device__ __forceinline__ unsigned int snx_pk2(float a, float b)
{
    bf16x2 t;
    t.x = (__bf16)a;
    t.y = (__bf16)b;
    return __builtin_bit_cast(unsigned int, t);
}

// ---------------------------------------------------------------------------
// Generic batched GEMM:  C[M,N] = scale * (A_bf16 x B_bf16) + bias (+gelu)
//   A: fp32 row-major (m,k) at A + m*lda + k    (optional conv row shift)
//   B: fp32 element (n,k) at B + n*sBN + k*sBK  (sBK==1 -> float4 fast path)
//   Batch z = (zb = z/nh2, zh = z%nh2) with independent strides.
//   Requirements (all call sites satisfy them): K % 32 == 0; fast-path
//   pointers/strides 16B aligned.
//
//   Block tile 128x64, 8 waves, each wave owns a 32x32 C tile:
//   2 A-frags x 2 B-frags -> 4 x v_wmma_f32_16x16x32_bf16 per K-step.
//   LDS rows padded to 80B so all fragment/staging accesses are b128.
//   Register-staged pipeline: next global tile issued before the WMMAs.
// ---------------------------------------------------------------------------
__global__ __launch_bounds__(256)
void snx_gemm_bf16_kernel(const float* __restrict__ A, long lda, long sAb, long sAh,
                          const float* __restrict__ Bp, long sBN, long sBK, long sBb, long sBh,
                          float* __restrict__ C, long ldc, long sCb, long sCh,
                          const float* __restrict__ bias,
                          int M, int N, int K, int nh2,
                          float scale, int gelu_flag, int accumulate,
                          int row_shift, int row_period)
{
    __shared__ __align__(16) unsigned int As_u[128][20];  // [m][k/2], 2xbf16 packed
    __shared__ __align__(16) unsigned int Bs_u[64][20];   // [n][k/2]

    const int z  = blockIdx.z;
    const int zb = z / nh2;
    const int zh = z % nh2;
    const float* Ab = A  + (long)zb * sAb + (long)zh * sAh;
    const float* Bb = Bp + (long)zb * sBb + (long)zh * sBh;
    float*       Cb = C  + (long)zb * sCb + (long)zh * sCh;

    const int m0 = blockIdx.y * 128;
    const int n0 = blockIdx.x * 64;

    const int tid  = threadIdx.x;
    const int lane = tid & 31;
    const int wave = tid >> 5;
    const int wm   = (wave & 3) * 32;   // wave's 32-row strip within 128
    const int wn   = (wave >> 2) * 32;  // wave's 32-col strip within 64

    // ---- staging geometry (k-independent) ----
    // A: thread covers row (tid>>1), 16 consecutive floats at (tid&1)*16
    const int arow = tid >> 1;
    const int akf  = (tid & 1) * 16;
    float amask = 1.0f;
    {
        int m  = m0 + arow;
        int mc = (m < M) ? m : (M - 1);
        if (m >= M) amask = 0.0f;
        long rofs = mc;
        if (row_period > 0) {                       // conv halo addressing
            int sbase = mc - (mc % row_period);
            int s     = (mc % row_period) + row_shift;
            if (s < 0)                { s = 0;               amask = 0.0f; }
            else if (s >= row_period) { s = row_period - 1;  amask = 0.0f; }
            rofs = (long)sbase + s;
        }
        Ab += rofs * lda + akf;                     // now Ab points at k=akf
    }
    // B: thread covers col (tid>>2), 8 consecutive k at (tid&3)*8
    const int brow = tid >> 2;
    const int bkf  = (tid & 3) * 8;
    float bmask = 1.0f;
    {
        int n  = n0 + brow;
        int nc = (n < N) ? n : (N - 1);
        if (n >= N) bmask = 0.0f;
        Bb += (long)nc * sBN;
    }
    const bool bfast = (sBK == 1);

    // ---- accumulators ----
    const int half = lane >> 4;
    const int l15  = lane & 15;
    const int cm0  = m0 + wm + half * 8;            // base row for a0 tiles
    const int cn0  = n0 + wn + l15;
    const int cn1  = cn0 + 16;

    v8f c00, c01, c10, c11;
    if (accumulate) {
#pragma unroll
        for (int v = 0; v < 8; ++v) {
            int ma = cm0 + v, mb = ma + 16;
            c00[v] = (ma < M && cn0 < N) ? Cb[(long)ma * ldc + cn0] : 0.0f;
            c01[v] = (ma < M && cn1 < N) ? Cb[(long)ma * ldc + cn1] : 0.0f;
            c10[v] = (mb < M && cn0 < N) ? Cb[(long)mb * ldc + cn0] : 0.0f;
            c11[v] = (mb < M && cn1 < N) ? Cb[(long)mb * ldc + cn1] : 0.0f;
        }
    } else {
#pragma unroll
        for (int v = 0; v < 8; ++v) { c00[v]=0.f; c01[v]=0.f; c10[v]=0.f; c11[v]=0.f; }
    }

    // ---- register-staged loaders (unconditional, vectorized where possible) --
    float4 ra0, ra1, ra2, ra3;      // 16 floats of A
    float4 rb0, rb1;                // 8 floats of B

    auto load_tiles = [&](int k0) {
        const float4* pa = (const float4*)(Ab + k0);
        ra0 = pa[0]; ra1 = pa[1]; ra2 = pa[2]; ra3 = pa[3];
        if (bfast) {
            const float4* pb = (const float4*)(Bb + k0 + bkf);
            rb0 = pb[0]; rb1 = pb[1];
        } else {
            long kk = (long)(k0 + bkf) * sBK;
            rb0.x = Bb[kk];            rb0.y = Bb[kk + sBK];
            rb0.z = Bb[kk + 2 * sBK];  rb0.w = Bb[kk + 3 * sBK];
            rb1.x = Bb[kk + 4 * sBK];  rb1.y = Bb[kk + 5 * sBK];
            rb1.z = Bb[kk + 6 * sBK];  rb1.w = Bb[kk + 7 * sBK];
        }
    };
    auto stage_tiles = [&]() {
        const int akp = akf >> 1;                   // k-pair base (0 or 8), 16B-aligned
        uint4 pa0, pa1;
        pa0.x = snx_pk2(ra0.x * amask, ra0.y * amask);
        pa0.y = snx_pk2(ra0.z * amask, ra0.w * amask);
        pa0.z = snx_pk2(ra1.x * amask, ra1.y * amask);
        pa0.w = snx_pk2(ra1.z * amask, ra1.w * amask);
        pa1.x = snx_pk2(ra2.x * amask, ra2.y * amask);
        pa1.y = snx_pk2(ra2.z * amask, ra2.w * amask);
        pa1.z = snx_pk2(ra3.x * amask, ra3.y * amask);
        pa1.w = snx_pk2(ra3.z * amask, ra3.w * amask);
        *(uint4*)&As_u[arow][akp]     = pa0;        // ds_store_b128
        *(uint4*)&As_u[arow][akp + 4] = pa1;        // ds_store_b128
        const int bkp = bkf >> 1;                   // 0,4,8,12 -> 16B-aligned
        uint4 pb;
        pb.x = snx_pk2(rb0.x * bmask, rb0.y * bmask);
        pb.y = snx_pk2(rb0.z * bmask, rb0.w * bmask);
        pb.z = snx_pk2(rb1.x * bmask, rb1.y * bmask);
        pb.w = snx_pk2(rb1.z * bmask, rb1.w * bmask);
        *(uint4*)&Bs_u[brow][bkp] = pb;             // ds_store_b128
    };

    // ---- main K loop (K % 32 == 0 at every call site) ----
    load_tiles(0);
    const int am0 = wm + l15;
    const int am1 = am0 + 16;
    const int bn0 = wn + l15;
    const int bn1 = bn0 + 16;
    for (int k0 = 0; k0 < K; k0 += 32) {
        stage_tiles();
        __syncthreads();
        if (k0 + 32 < K) load_tiles(k0 + 32);   // issue next tile; latency hides under WMMA

        // Fragment gathers: A and B u[0..3]/u[4..7] are contiguous
        // 16B-aligned runs -> ds_load_b128 pairs.
        SnxFrag fa0, fa1, fb0, fb1;
        fa0.q[0] = *(const uint4*)&As_u[am0][half * 4];
        fa0.q[1] = *(const uint4*)&As_u[am0][8 + half * 4];
        fa1.q[0] = *(const uint4*)&As_u[am1][half * 4];
        fa1.q[1] = *(const uint4*)&As_u[am1][8 + half * 4];
        fb0.q[0] = *(const uint4*)&Bs_u[bn0][half * 4];
        fb0.q[1] = *(const uint4*)&Bs_u[bn0][8 + half * 4];
        fb1.q[0] = *(const uint4*)&Bs_u[bn1][half * 4];
        fb1.q[1] = *(const uint4*)&Bs_u[bn1][8 + half * 4];

        c00 = __builtin_amdgcn_wmma_f32_16x16x32_bf16(false, fa0.v, false, fb0.v,
                                                      (short)0, c00, false, false);
        c01 = __builtin_amdgcn_wmma_f32_16x16x32_bf16(false, fa0.v, false, fb1.v,
                                                      (short)0, c01, false, false);
        c10 = __builtin_amdgcn_wmma_f32_16x16x32_bf16(false, fa1.v, false, fb0.v,
                                                      (short)0, c10, false, false);
        c11 = __builtin_amdgcn_wmma_f32_16x16x32_bf16(false, fa1.v, false, fb1.v,
                                                      (short)0, c11, false, false);
        __syncthreads();
    }

    // ---- epilogue: scale, bias, gelu ----
    auto emit = [&](int m, int n, float acc) {
        if (m < M && n < N) {
            float val = acc * scale + (bias ? bias[n] : 0.0f);
            if (gelu_flag) val = 0.5f * val * (1.0f + erff(val * 0.70710678f));
            Cb[(long)m * ldc + n] = val;
        }
    };
#pragma unroll
    for (int v = 0; v < 8; ++v) {
        int ma = cm0 + v, mb = ma + 16;
        emit(ma, cn0, c00[v]);
        emit(ma, cn1, c01[v]);
        emit(mb, cn0, c10[v]);
        emit(mb, cn1, c11[v]);
    }
}

// ---------------------------------------------------------------------------
// Elementwise / scan helper kernels
// ---------------------------------------------------------------------------
__global__ void snx_embed_kernel(const int* __restrict__ ids,
                                 const float* __restrict__ emb,
                                 const float* __restrict__ pos,
                                 float* __restrict__ x, long total)
{
    long i = (long)blockIdx.x * blockDim.x + threadIdx.x;
    if (i >= total) return;
    long row = i / Hc;
    int  h   = (int)(i % Hc);
    int  s   = (int)(row % Sc);
    x[i] = emb[(long)ids[row] * Hc + h] + pos[(long)s * Hc + h];
}

// x[row] = LayerNorm(x[row] + h[row]) * g + b   (row length == 256 == blockDim)
__global__ __launch_bounds__(256)
void snx_ln_res_kernel(float* __restrict__ x, const float* __restrict__ hbuf,
                       const float* __restrict__ g, const float* __restrict__ b)
{
    __shared__ float red[256];
    long row = blockIdx.x;
    int  t   = threadIdx.x;
    float v  = x[row * Hc + t] + hbuf[row * Hc + t];
    red[t] = v; __syncthreads();
    for (int s = 128; s > 0; s >>= 1) { if (t < s) red[t] += red[t + s]; __syncthreads(); }
    float mean = red[0] * (1.0f / Hc); __syncthreads();
    float d = v - mean;
    red[t] = d * d; __syncthreads();
    for (int s = 128; s > 0; s >>= 1) { if (t < s) red[t] += red[t + s]; __syncthreads(); }
    float var = red[0] * (1.0f / Hc);
    x[row * Hc + t] = d * rsqrtf(var + 1e-5f) * g[t] + b[t];
}

__global__ __launch_bounds__(256)
void snx_softmax_kernel(float* __restrict__ x, int L)
{
    __shared__ float red[256];
    long  row = blockIdx.x;
    float* p  = x + row * (long)L;
    int   t   = threadIdx.x;
    float mx  = -3.4e38f;
    for (int i = t; i < L; i += 256) mx = fmaxf(mx, p[i]);
    red[t] = mx; __syncthreads();
    for (int s = 128; s > 0; s >>= 1) { if (t < s) red[t] = fmaxf(red[t], red[t + s]); __syncthreads(); }
    mx = red[0]; __syncthreads();
    float sum = 0.0f;
    for (int i = t; i < L; i += 256) { float e = expf(p[i] - mx); p[i] = e; sum += e; }
    red[t] = sum; __syncthreads();
    for (int s = 128; s > 0; s >>= 1) { if (t < s) red[t] += red[t + s]; __syncthreads(); }
    float inv = 1.0f / red[0];
    for (int i = t; i < L; i += 256) p[i] *= inv;
}

// LIF scan over the sequence dim, in place: proj -> spikes. One thread per (b,h).
__global__ void snx_lif_kernel(float* __restrict__ proj)
{
    int idx = blockIdx.x * blockDim.x + threadIdx.x;
    if (idx >= Bc * Hc) return;
    int b = idx / Hc, h = idx % Hc;
    float* p = proj + (long)b * Sc * Hc + h;
    float mem = 0.0f;
    for (int s = 0; s < Sc; ++s) {
        float xt    = p[(long)s * Hc];
        float reset = (mem - 1.0f > 0.0f) ? 1.0f : 0.0f;
        mem = 0.9f * mem + xt - reset;                 // subtract-reset, THR=1
        p[(long)s * Hc] = (mem - 1.0f > 0.0f) ? 1.0f : 0.0f;
    }
}

__global__ void snx_concat_kernel(const float* __restrict__ lg, const float* __restrict__ mt,
                                  const float* __restrict__ ln, const float* __restrict__ pl,
                                  float* __restrict__ out, long total)
{
    long i = (long)blockIdx.x * blockDim.x + threadIdx.x;
    if (i >= total) return;
    long row = i / (4 * Hc);
    int  c   = (int)(i % (4 * Hc));
    int  sel = c / Hc, j = c % Hc;
    long src = row * Hc + j;
    float v;
    if      (sel == 0) v = lg[src];
    else if (sel == 1) v = mt[src];
    else if (sel == 2) v = ln[src];
    else               v = pl[src];
    out[i] = v;
}

__global__ void snx_mean_kernel(const float* __restrict__ x, float* __restrict__ out)
{
    int idx = blockIdx.x * blockDim.x + threadIdx.x;   // b*H + h
    if (idx >= Bc * Hc) return;
    int b = idx / Hc, h = idx % Hc;
    float s = 0.0f;
    for (int t = 0; t < Sc; ++t) s += x[(long)b * Sc * Hc + (long)t * Hc + h];
    out[idx] = s * (1.0f / Sc);
}

__global__ void snx_readw_kernel(const float* __restrict__ pm, const float* __restrict__ rw,
                                 const float* __restrict__ rb, float* __restrict__ out)
{
    int idx = blockIdx.x * blockDim.x + threadIdx.x;   // b*M + m
    if (idx >= Bc * Mc) return;
    int b = idx / Mc, m = idx % Mc;
    float s = 0.0f;
    for (int k = 0; k < Hc; ++k) s += pm[b * Hc + k] * rw[m * Hc + k];
    out[idx] = s + rb[m];
}

__global__ void snx_memread_kernel(const float* __restrict__ w, const float* __restrict__ mem,
                                   float* __restrict__ out)
{
    int idx = blockIdx.x * blockDim.x + threadIdx.x;   // b*H + h
    if (idx >= Bc * Hc) return;
    int b = idx / Hc, h = idx % Hc;
    float s = 0.0f;
    for (int m = 0; m < Mc; ++m) s += w[b * Mc + m] * mem[(long)m * Hc + h];
    out[idx] = s;
}

// ---------------------------------------------------------------------------
// Host side
// ---------------------------------------------------------------------------
static void snx_gemm(hipStream_t st,
                     const float* A, long lda, long sAb, long sAh,
                     const float* Bp, long sBN, long sBK, long sBb, long sBh,
                     float* C, long ldc, long sCb, long sCh,
                     const float* bias, int M, int N, int K,
                     int nh2, int batches, float scale,
                     int gelu, int acc, int rshift, int rperiod)
{
    dim3 grid((N + 63) / 64, (M + 127) / 128, batches);
    snx_gemm_bf16_kernel<<<grid, dim3(256), 0, st>>>(A, lda, sAb, sAh,
                                                     Bp, sBN, sBK, sBb, sBh,
                                                     C, ldc, sCb, sCh,
                                                     bias, M, N, K, nh2,
                                                     scale, gelu, acc, rshift, rperiod);
}

extern "C" void kernel_launch(void* const* d_in, const int* in_sizes, int n_in,
                              void* d_out, int out_size, void* d_ws, size_t ws_size,
                              hipStream_t stream)
{
    (void)in_sizes; (void)n_in; (void)out_size; (void)ws_size;

    const int*   ids    = (const int*)  d_in[0];
    const float* embed  = (const float*)d_in[1];
    const float* pos    = (const float*)d_in[2];
    const float* qkv_w  = (const float*)d_in[3];
    const float* qkv_b  = (const float*)d_in[4];
    const float* out_w  = (const float*)d_in[5];
    const float* out_b  = (const float*)d_in[6];
    const float* ln1_g  = (const float*)d_in[7];
    const float* ln1_b  = (const float*)d_in[8];
    const float* ff1_w  = (const float*)d_in[9];
    const float* ff1_b  = (const float*)d_in[10];
    const float* ff2_w  = (const float*)d_in[11];
    const float* ff2_b  = (const float*)d_in[12];
    const float* ln2_g  = (const float*)d_in[13];
    const float* ln2_b  = (const float*)d_in[14];
    const float* log_w  = (const float*)d_in[15];
    const float* log_b  = (const float*)d_in[16];
    const float* mat_w  = (const float*)d_in[17];
    const float* mat_b  = (const float*)d_in[18];
    const float* conv_w = (const float*)d_in[19];
    const float* conv_b = (const float*)d_in[20];
    const float* read_w = (const float*)d_in[21];
    const float* read_b = (const float*)d_in[22];
    const float* memory = (const float*)d_in[23];
    const float* head_w = (const float*)d_in[24];
    const float* head_b = (const float*)d_in[25];

    float* logits   = (float*)d_out;                        // [B,S,V]
    float* mem_read = logits + (long)Bc * Sc * Vc;          // [B,H]

    // Workspace arena (floats), with aliasing after the transformer layers.
    float* ws = (float*)d_ws;
    const long N_XSH  = (long)Bc * Sc * Hc;     // 2,097,152
    float* x     = ws;                          // [B,S,H]   plan
    float* qkv   = x    + N_XSH;                // [B,S,3H]  (later: logico/matem/ling)
    float* att   = qkv  + 3 * N_XSH;            // [B,NH,S,S] (later: combined [B,S,4H])
    float* tmp   = att  + (long)Bc * NHc * Sc * Sc;  // [B,S,H]  attention out
    float* tmp2  = tmp  + N_XSH;                // [B,S,H]  proj before ln_res
    float* ff1o  = tmp2 + N_XSH;                // [B,S,FF]
    float* small = ff1o + (long)Bc * Sc * FFc;  // plan_mean [B,H] + read_logits [B,M]
    float* pmean = small;
    float* rlog  = small + Bc * Hc;

    const long rows = (long)Bc * Sc;            // 8192

    // 0) embedding + positional
    {
        long total = rows * Hc;
        snx_embed_kernel<<<dim3((unsigned)((total + 255) / 256)), dim3(256), 0, stream>>>(
            ids, embed, pos, x, total);
    }

    // 1) transformer layers
    for (int L = 0; L < NLc; ++L) {
        const float* qw  = qkv_w + (long)L * 3 * Hc * Hc;
        const float* qb  = qkv_b + (long)L * 3 * Hc;
        const float* ow  = out_w + (long)L * Hc * Hc;
        const float* ob  = out_b + (long)L * Hc;
        const float* f1w = ff1_w + (long)L * FFc * Hc;
        const float* f1b = ff1_b + (long)L * FFc;
        const float* f2w = ff2_w + (long)L * Hc * FFc;
        const float* f2b = ff2_b + (long)L * Hc;

        // qkv = x @ qkv_w^T + qkv_b   [8192 x 768], K=256
        snx_gemm(stream, x, Hc, 0, 0,
                 qw, Hc, 1, 0, 0,
                 qkv, 3 * Hc, 0, 0,
                 qb, (int)rows, 3 * Hc, Hc, 1, 1, 1.0f, 0, 0, 0, 0);

        // scores = q k^T / sqrt(hd)   batched over (b,h): [512 x 512], K=64
        snx_gemm(stream, qkv, 3 * Hc, (long)Sc * 3 * Hc, HDc,
                 qkv + Hc, 3 * Hc, 1, (long)Sc * 3 * Hc, HDc,
                 att, Sc, (long)NHc * Sc * Sc, (long)Sc * Sc,
                 nullptr, Sc, Sc, HDc, NHc, Bc * NHc, 0.125f, 0, 0, 0, 0);

        // softmax over rows of length 512
        snx_softmax_kernel<<<dim3((unsigned)(Bc * NHc * Sc)), dim3(256), 0, stream>>>(att, Sc);

        // o = att @ v   batched: [512 x 64], K=512   (v strided-K slow path)
        snx_gemm(stream, att, Sc, (long)NHc * Sc * Sc, (long)Sc * Sc,
                 qkv + 2 * Hc, 1, 3 * Hc, (long)Sc * 3 * Hc, HDc,
                 tmp, Hc, (long)Sc * Hc, HDc,
                 nullptr, Sc, HDc, Sc, NHc, Bc * NHc, 1.0f, 0, 0, 0, 0);

        // proj = o @ out_w^T + out_b
        snx_gemm(stream, tmp, Hc, 0, 0,
                 ow, Hc, 1, 0, 0,
                 tmp2, Hc, 0, 0,
                 ob, (int)rows, Hc, Hc, 1, 1, 1.0f, 0, 0, 0, 0);

        // x = LN(x + proj)
        snx_ln_res_kernel<<<dim3((unsigned)rows), dim3(256), 0, stream>>>(
            x, tmp2, ln1_g + L * Hc, ln1_b + L * Hc);

        // ff1 = gelu(x @ ff1_w^T + b)   [8192 x 512], K=256
        snx_gemm(stream, x, Hc, 0, 0,
                 f1w, Hc, 1, 0, 0,
                 ff1o, FFc, 0, 0,
                 f1b, (int)rows, FFc, Hc, 1, 1, 1.0f, 1, 0, 0, 0);

        // ff2 = ff1 @ ff2_w^T + b   [8192 x 256], K=512
        snx_gemm(stream, ff1o, FFc, 0, 0,
                 f2w, FFc, 1, 0, 0,
                 tmp2, Hc, 0, 0,
                 f2b, (int)rows, Hc, FFc, 1, 1, 1.0f, 0, 0, 0, 0);

        // x = LN(x + ff2)
        snx_ln_res_kernel<<<dim3((unsigned)rows), dim3(256), 0, stream>>>(
            x, tmp2, ln2_g + L * Hc, ln2_b + L * Hc);
    }

    // x == plan. Spiking / conv branches (alias the qkv region).
    float* logico   = qkv;
    float* matem    = qkv + N_XSH;
    float* ling     = qkv + 2 * N_XSH;
    float* combined = att;                      // [B,S,4H] aliases scores buffer

    // logico / matem projections then LIF scans (in place)
    snx_gemm(stream, x, Hc, 0, 0, log_w, Hc, 1, 0, 0,
             logico, Hc, 0, 0, log_b, (int)rows, Hc, Hc, 1, 1, 1.0f, 0, 0, 0, 0);
    snx_gemm(stream, x, Hc, 0, 0, mat_w, Hc, 1, 0, 0,
             matem, Hc, 0, 0, mat_b, (int)rows, Hc, Hc, 1, 1, 1.0f, 0, 0, 0, 0);
    snx_lif_kernel<<<dim3((Bc * Hc + 255) / 256), dim3(256), 0, stream>>>(logico);
    snx_lif_kernel<<<dim3((Bc * Hc + 255) / 256), dim3(256), 0, stream>>>(matem);

    // conv1d (kernel 3, pad 1) as 3 accumulated GEMMs with shifted A rows
    for (int t = 0; t < 3; ++t) {
        snx_gemm(stream, x, Hc, 0, 0,
                 conv_w + t, 3 * Hc, 3, 0, 0,
                 ling, Hc, 0, 0,
                 (t == 0) ? conv_b : nullptr,
                 (int)rows, Hc, Hc, 1, 1, 1.0f, 0, /*acc=*/(t > 0),
                 /*row_shift=*/t - 1, /*row_period=*/Sc);
    }

    // combined = concat([logico, matem, ling, plan], -1)
    {
        long total = rows * 4 * Hc;
        snx_concat_kernel<<<dim3((unsigned)((total + 255) / 256)), dim3(256), 0, stream>>>(
            logico, matem, ling, x, combined, total);
    }

    // plan_mean -> read logits -> softmax -> memory_read
    snx_mean_kernel<<<dim3((Bc * Hc + 255) / 256), dim3(256), 0, stream>>>(x, pmean);
    snx_readw_kernel<<<dim3((Bc * Mc + 255) / 256), dim3(256), 0, stream>>>(pmean, read_w, read_b, rlog);
    snx_softmax_kernel<<<dim3(Bc), dim3(256), 0, stream>>>(rlog, Mc);
    snx_memread_kernel<<<dim3((Bc * Hc + 255) / 256), dim3(256), 0, stream>>>(rlog, memory, mem_read);

    // logits = combined @ head_w^T + head_b   [8192 x 8000], K=1024 (dominant GEMM)
    snx_gemm(stream, combined, 4 * Hc, 0, 0,
             head_w, 4 * Hc, 1, 0, 0,
             logits, Vc, 0, 0,
             head_b, (int)rows, Vc, 4 * Hc, 1, 1, 1.0f, 0, 0, 0, 0);
}
